// MoEBlock_36953898615263
// MI455X (gfx1250) — matrix operand order repaired and checked by reading, not compiled, not run on hardware
//
#include <hip/hip_runtime.h>

// ---------------------------------------------------------------------------
// MoE block (top-1 routing, shared wi/wo, rank-4 per-expert LoRA) for gfx1250.
// B=4, S=2048, D=1024, F=4096, E=8, R=4.  bf16 WMMA (f32 accum) GEMMs with
// async global->LDS double-buffered staging.
// ---------------------------------------------------------------------------

#define NT 8192     // B*S tokens
#define DD 1024
#define FF 4096
#define EE 8
#define RR 4

typedef __attribute__((ext_vector_type(16))) __bf16 v16bf;
typedef __attribute__((ext_vector_type(8)))  float  v8f;
typedef int v4i __attribute__((ext_vector_type(4)));

union FragU {
    v16bf v;
    uint4 q[2];
};

// ------------------------- async global->LDS path --------------------------
#if defined(__has_builtin)
#if __has_builtin(__builtin_amdgcn_global_load_async_to_lds_b128)
#define HAVE_ASYNC_LDS 1
#endif
#endif
#ifndef HAVE_ASYNC_LDS
#define HAVE_ASYNC_LDS 0
#endif

typedef __attribute__((address_space(1))) v4i* as1_v4i_ptr;  // global int4*
typedef __attribute__((address_space(3))) v4i* as3_v4i_ptr;  // LDS int4*

static __device__ inline void async_copy16(const void* g, void* l) {
#if HAVE_ASYNC_LDS
    __builtin_amdgcn_global_load_async_to_lds_b128((as1_v4i_ptr)g,
                                                   (as3_v4i_ptr)l, 0, 0);
#else
    *(uint4*)l = *(const uint4*)g;
#endif
}

#if HAVE_ASYNC_LDS
#if __has_builtin(__builtin_amdgcn_s_wait_asynccnt)
#define WAIT_ASYNC(n) __builtin_amdgcn_s_wait_asynccnt(n)
#else
#define WAIT_ASYNC(n) asm volatile("s_wait_asynccnt " #n ::: "memory")
#endif
#else
#define WAIT_ASYNC(n) ((void)0)
#endif

// ---------------------------------------------------------------------------
static __device__ inline unsigned short f2bf(float f) {
    unsigned int u = __float_as_uint(f);
    u += 0x7FFFu + ((u >> 16) & 1u);      // round-to-nearest-even
    return (unsigned short)(u >> 16);
}

static __device__ inline v8f wmma_bf16(v16bf a, v16bf b, v8f c) {
    return __builtin_amdgcn_wmma_f32_16x16x32_bf16(
        /*neg_a=*/false, a, /*neg_b=*/false, b,
        /*c_mod=*/(short)0, c, /*reuse_a=*/false, /*reuse_b=*/false);
}

// A-fragment (16x32 bf16) from LDS tile As[128][40] (row-major, ushort).
// lanes 0-15 (h=0): V0..3 = K 0..7, V4..7 = K 16..23
// lanes16-31 (h=1): V0..3 = K 8..15, V4..7 = K 24..31
static __device__ inline v16bf load_a_frag(const unsigned short* As,
                                           int rowBase, int lane) {
    int h = (lane >> 4) & 1;
    int m = lane & 15;
    const unsigned short* p = As + (rowBase + m) * 40;
    FragU f;
    f.q[0] = *(const uint4*)(p + 8 * h);
    f.q[1] = *(const uint4*)(p + 16 + 8 * h);
    return f.v;
}

// B-fragment (32x16 bf16) from LDS tile Bs[32][136]: lane l holds col N=l%16,
// same K order as A: h=0: K 0..7,16..23; h=1: K 8..15,24..31.
static __device__ inline v16bf load_b_frag(const unsigned short* Bs,
                                           int colBase, int lane) {
    int h = (lane >> 4) & 1;
    const unsigned short* p = Bs + colBase + (lane & 15);
    FragU f;
    unsigned short* u = (unsigned short*)f.q;
#pragma unroll
    for (int s = 0; s < 8; ++s) {
        u[s] = p[(8 * h + s) * 136];
        u[8 + s] = p[(16 + 8 * h + s) * 136];
    }
    return f.v;
}

// ---------------------------------------------------------------------------
// Prep: fp32 -> bf16 conversions (with weight transposes so B[k][n] is linear)
// ---------------------------------------------------------------------------
__global__ void cvt_x_kernel(const float* __restrict__ s,
                             unsigned short* __restrict__ d, int n) {
    for (int i = blockIdx.x * blockDim.x + threadIdx.x; i < n;
         i += gridDim.x * blockDim.x)
        d[i] = f2bf(s[i]);
}

// wi_W [F][D] -> wibT [D][F]
__global__ void cvt_wiT_kernel(const float* __restrict__ wi,
                               unsigned short* __restrict__ wibT) {
    int idx = blockIdx.x * blockDim.x + threadIdx.x;   // over F*D
    int f = idx >> 10;
    int dcol = idx & 1023;
    wibT[(size_t)dcol * FF + f] = f2bf(wi[idx]);
}

// wo_W [D][F] -> wobT [F][D]
__global__ void cvt_woT_kernel(const float* __restrict__ wo,
                               unsigned short* __restrict__ wobT) {
    int idx = blockIdx.x * blockDim.x + threadIdx.x;   // over D*F
    int dcol = idx >> 12;
    int f = idx & 4095;
    wobT[(size_t)f * DD + dcol] = f2bf(wo[idx]);
}

// ---------------------------------------------------------------------------
// Router: top1[t] = argmax_e (x[t]·gate_W[e] + gate_b[e]);
//         lora_r[t][r] = x[t]·lora_A[top1[t]][r]
// ---------------------------------------------------------------------------
__global__ __launch_bounds__(128) void router_kernel(
    const float* __restrict__ x, const float* __restrict__ gW,
    const float* __restrict__ gb, const float* __restrict__ lA,
    int* __restrict__ top1, float* __restrict__ lr) {
    int t = blockIdx.x;
    const float* xr = x + (size_t)t * DD;
    __shared__ float red[128];
    __shared__ int se;
    float best = -3.4e38f;
    int bi = 0;
    for (int e = 0; e < EE; ++e) {
        float p = 0.f;
        const float* w = gW + (size_t)e * DD;
        for (int d = threadIdx.x; d < DD; d += 128) p += xr[d] * w[d];
        red[threadIdx.x] = p;
        __syncthreads();
        for (int s = 64; s > 0; s >>= 1) {
            if (threadIdx.x < (unsigned)s) red[threadIdx.x] += red[threadIdx.x + s];
            __syncthreads();
        }
        if (threadIdx.x == 0) {
            float L = red[0] + gb[e];
            if (L > best) { best = L; bi = e; }
        }
        __syncthreads();
    }
    if (threadIdx.x == 0) { top1[t] = bi; se = bi; }
    __syncthreads();
    int e = se;
    for (int r = 0; r < RR; ++r) {
        float p = 0.f;
        const float* a = lA + ((size_t)e * RR + r) * DD;
        for (int d = threadIdx.x; d < DD; d += 128) p += xr[d] * a[d];
        red[threadIdx.x] = p;
        __syncthreads();
        for (int s = 64; s > 0; s >>= 1) {
            if (threadIdx.x < (unsigned)s) red[threadIdx.x] += red[threadIdx.x + s];
            __syncthreads();
        }
        if (threadIdx.x == 0) lr[t * RR + r] = red[0];
        __syncthreads();
    }
}

// ---------------------------------------------------------------------------
// GEMM1: inter[t][f] = relu( x@wiT + wi_b[f] + r[t]·lora_B[top1[t]][f][:] )
// M=8192 N=4096 K=1024. 128x128 tile, KC=32 double-buffered async staging,
// 8 waves of 32x64 (2x4 WMMA fragments each).
// ---------------------------------------------------------------------------
__global__ __launch_bounds__(256) void moe_gemm1(
    const unsigned short* __restrict__ xb,    // [NT][DD] bf16
    const unsigned short* __restrict__ wibT,  // [DD][FF] bf16
    const float* __restrict__ wi_b,           // [FF]
    const int* __restrict__ top1,             // [NT]
    const float* __restrict__ lora_r,         // [NT][RR]
    const float* __restrict__ lora_B,         // [EE][FF][RR]
    unsigned short* __restrict__ inter)       // [NT][FF] bf16
{
    __shared__ unsigned short As[2][128 * 40];
    __shared__ unsigned short Bs[2][32 * 136];
    const int tilesN = FF / 128;                // 32
    int bm = blockIdx.x / tilesN;
    int bn = blockIdx.x % tilesN;
    int row0 = bm * 128, col0 = bn * 128;
    int tid = threadIdx.x;
    int wave = tid >> 5, lane = tid & 31;
    int wm = wave >> 1, wn = wave & 1;          // 4x2 wave grid

    v8f acc[2][4];
#pragma unroll
    for (int i = 0; i < 2; ++i)
#pragma unroll
        for (int j = 0; j < 4; ++j)
#pragma unroll
            for (int v = 0; v < 8; ++v) acc[i][j][v] = 0.f;

    // Issue async copy of one KC=32 chunk (A: 128x32, B: 32x128) -> buffer.
    auto issue = [&](int ci, int buf) {
#pragma unroll
        for (int it = 0; it < 2; ++it) {
            int idx = tid + 256 * it;           // 0..511 (512 x 16B each)
            int ra = idx >> 2, ca = (idx & 3) << 3;
            async_copy16(xb + (size_t)(row0 + ra) * DD + ci * 32 + ca,
                         &As[buf][ra * 40 + ca]);
            int rb = idx >> 4, cb = (idx & 15) << 3;
            async_copy16(wibT + (size_t)(ci * 32 + rb) * FF + col0 + cb,
                         &Bs[buf][rb * 136 + cb]);
        }
    };

    const int NK = DD / 32;                     // 32 chunks
    issue(0, 0);
    for (int i = 0; i < NK; ++i) {
        int buf = i & 1;
        if (i + 1 < NK) {
            issue(i + 1, buf ^ 1);
            WAIT_ASYNC(4);                      // newest 4 = next chunk
        } else {
            WAIT_ASYNC(0);
        }
        __syncthreads();
        v16bf a0 = load_a_frag(As[buf], wm * 32 + 0,  lane);
        v16bf a1 = load_a_frag(As[buf], wm * 32 + 16, lane);
        v16bf b0 = load_b_frag(Bs[buf], wn * 64 + 0,  lane);
        v16bf b1 = load_b_frag(Bs[buf], wn * 64 + 16, lane);
        v16bf b2 = load_b_frag(Bs[buf], wn * 64 + 32, lane);
        v16bf b3 = load_b_frag(Bs[buf], wn * 64 + 48, lane);
        acc[0][0] = wmma_bf16(a0, b0, acc[0][0]);
        acc[0][1] = wmma_bf16(a0, b1, acc[0][1]);
        acc[0][2] = wmma_bf16(a0, b2, acc[0][2]);
        acc[0][3] = wmma_bf16(a0, b3, acc[0][3]);
        acc[1][0] = wmma_bf16(a1, b0, acc[1][0]);
        acc[1][1] = wmma_bf16(a1, b1, acc[1][1]);
        acc[1][2] = wmma_bf16(a1, b2, acc[1][2]);
        acc[1][3] = wmma_bf16(a1, b3, acc[1][3]);
        __syncthreads();
    }
    // epilogue: + wi_b + LoRA(top1), relu, bf16 store
    int nlo = lane & 15;
    int mhi = (lane >> 4) << 3;
#pragma unroll
    for (int i = 0; i < 2; ++i) {
#pragma unroll
        for (int v = 0; v < 8; ++v) {
            int t = row0 + wm * 32 + i * 16 + mhi + v;
            int e = top1[t];
            float4 rr = *(const float4*)(lora_r + t * RR);
#pragma unroll
            for (int j = 0; j < 4; ++j) {
                int f = col0 + wn * 64 + j * 16 + nlo;
                float val = acc[i][j][v] + wi_b[f];
                float4 lb = *(const float4*)(lora_B + ((size_t)e * FF + f) * RR);
                val += rr.x * lb.x + rr.y * lb.y + rr.z * lb.z + rr.w * lb.w;
                val = fmaxf(val, 0.f);
                inter[(size_t)t * FF + f] = f2bf(val);
            }
        }
    }
}

// ---------------------------------------------------------------------------
// GEMM2: out[t][d] = inter[t][:]·wobT[:][d] + wo_b[d]
// M=8192 N=1024 K=4096.
// ---------------------------------------------------------------------------
__global__ __launch_bounds__(256) void moe_gemm2(
    const unsigned short* __restrict__ inter,  // [NT][FF] bf16
    const unsigned short* __restrict__ wobT,   // [FF][DD] bf16
    const float* __restrict__ wo_b,            // [DD]
    float* __restrict__ out)                   // [NT][DD] fp32
{
    __shared__ unsigned short As[2][128 * 40];
    __shared__ unsigned short Bs[2][32 * 136];
    const int tilesN = DD / 128;                // 8
    int bm = blockIdx.x / tilesN;
    int bn = blockIdx.x % tilesN;
    int row0 = bm * 128, col0 = bn * 128;
    int tid = threadIdx.x;
    int wave = tid >> 5, lane = tid & 31;
    int wm = wave >> 1, wn = wave & 1;

    v8f acc[2][4];
#pragma unroll
    for (int i = 0; i < 2; ++i)
#pragma unroll
        for (int j = 0; j < 4; ++j)
#pragma unroll
            for (int v = 0; v < 8; ++v) acc[i][j][v] = 0.f;

    auto issue = [&](int ci, int buf) {
#pragma unroll
        for (int it = 0; it < 2; ++it) {
            int idx = tid + 256 * it;
            int ra = idx >> 2, ca = (idx & 3) << 3;
            async_copy16(inter + (size_t)(row0 + ra) * FF + ci * 32 + ca,
                         &As[buf][ra * 40 + ca]);
            int rb = idx >> 4, cb = (idx & 15) << 3;
            async_copy16(wobT + (size_t)(ci * 32 + rb) * DD + col0 + cb,
                         &Bs[buf][rb * 136 + cb]);
        }
    };

    const int NK = FF / 32;                     // 128 chunks
    issue(0, 0);
    for (int i = 0; i < NK; ++i) {
        int buf = i & 1;
        if (i + 1 < NK) {
            issue(i + 1, buf ^ 1);
            WAIT_ASYNC(4);
        } else {
            WAIT_ASYNC(0);
        }
        __syncthreads();
        v16bf a0 = load_a_frag(As[buf], wm * 32 + 0,  lane);
        v16bf a1 = load_a_frag(As[buf], wm * 32 + 16, lane);
        v16bf b0 = load_b_frag(Bs[buf], wn * 64 + 0,  lane);
        v16bf b1 = load_b_frag(Bs[buf], wn * 64 + 16, lane);
        v16bf b2 = load_b_frag(Bs[buf], wn * 64 + 32, lane);
        v16bf b3 = load_b_frag(Bs[buf], wn * 64 + 48, lane);
        acc[0][0] = wmma_bf16(a0, b0, acc[0][0]);
        acc[0][1] = wmma_bf16(a0, b1, acc[0][1]);
        acc[0][2] = wmma_bf16(a0, b2, acc[0][2]);
        acc[0][3] = wmma_bf16(a0, b3, acc[0][3]);
        acc[1][0] = wmma_bf16(a1, b0, acc[1][0]);
        acc[1][1] = wmma_bf16(a1, b1, acc[1][1]);
        acc[1][2] = wmma_bf16(a1, b2, acc[1][2]);
        acc[1][3] = wmma_bf16(a1, b3, acc[1][3]);
        __syncthreads();
    }
    int nlo = lane & 15;
    int mhi = (lane >> 4) << 3;
#pragma unroll
    for (int i = 0; i < 2; ++i) {
#pragma unroll
        for (int v = 0; v < 8; ++v) {
            int t = row0 + wm * 32 + i * 16 + mhi + v;
#pragma unroll
            for (int j = 0; j < 4; ++j) {
                int dcol = col0 + wn * 64 + j * 16 + nlo;
                out[(size_t)t * DD + dcol] = acc[i][j][v] + wo_b[dcol];
            }
        }
    }
}

// ---------------------------------------------------------------------------
// Host-side launcher
// ---------------------------------------------------------------------------
extern "C" void kernel_launch(void* const* d_in, const int* in_sizes, int n_in,
                              void* d_out, int out_size, void* d_ws, size_t ws_size,
                              hipStream_t stream) {
    const float* x   = (const float*)d_in[0];  // [B,S,D]
    const float* gW  = (const float*)d_in[1];  // [E,D]
    const float* gb  = (const float*)d_in[2];  // [E]
    const float* wiW = (const float*)d_in[3];  // [F,D]
    const float* wib = (const float*)d_in[4];  // [F]
    const float* woW = (const float*)d_in[5];  // [D,F]
    const float* wob = (const float*)d_in[6];  // [D]
    const float* lA  = (const float*)d_in[7];  // [E,R,D]
    const float* lB  = (const float*)d_in[8];  // [E,F,R]
    float* out = (float*)d_out;

    char* ws = (char*)d_ws;
    const size_t OFF_TOP1  = 0;                         // 32 KB
    const size_t OFF_LR    = OFF_TOP1 + (size_t)NT * 4; // 128 KB
    const size_t OFF_XB    = 163840;                    // 16 MB
    const size_t OFF_WIBT  = OFF_XB   + (size_t)NT * DD * 2;
    const size_t OFF_WOBT  = OFF_WIBT + (size_t)DD * FF * 2;
    const size_t OFF_INTER = OFF_WOBT + (size_t)FF * DD * 2;

    int*            top1   = (int*)(ws + OFF_TOP1);
    float*          lora_r = (float*)(ws + OFF_LR);
    unsigned short* xb     = (unsigned short*)(ws + OFF_XB);
    unsigned short* wibT   = (unsigned short*)(ws + OFF_WIBT);
    unsigned short* wobT   = (unsigned short*)(ws + OFF_WOBT);
    unsigned short* inter  = (unsigned short*)(ws + OFF_INTER);

    cvt_x_kernel  <<<4096, 256, 0, stream>>>(x, xb, NT * DD);
    cvt_wiT_kernel<<<(FF * DD) / 256, 256, 0, stream>>>(wiW, wibT);
    cvt_woT_kernel<<<(DD * FF) / 256, 256, 0, stream>>>(woW, wobT);
    router_kernel <<<NT, 128, 0, stream>>>(x, gW, gb, lA, top1, lora_r);

    moe_gemm1<<<(NT / 128) * (FF / 128), 256, 0, stream>>>(
        xb, wibT, wib, top1, lora_r, lB, inter);
    moe_gemm2<<<(NT / 128) * (DD / 128), 256, 0, stream>>>(
        inter, wobT, wob, out);
}
